// PALMPartialAttention_56942676410744
// MI455X (gfx1250) — compile-verified, matrix operand order checked by reading, not run on hardware
//
#include <hip/hip_runtime.h>
#include <hip/hip_bf16.h>
#include <stdint.h>

// ---------------------------------------------------------------------------
// PALM partial (cross) attention for MI455X / gfx1250, bf16 WMMA pipeline.
// B=4, T=2048, H=1024, NH=16, DK=64.
// ---------------------------------------------------------------------------

typedef __bf16 bf16;
typedef __attribute__((ext_vector_type(16))) __bf16 v16bf;
typedef __attribute__((ext_vector_type(4)))  __bf16 v4bf;
typedef __attribute__((ext_vector_type(8)))  float  v8f;

static constexpr int Bc = 4;
static constexpr int Tc = 2048;
static constexpr int Hc = 1024;
static constexpr int NHc = 16;
static constexpr int DKc = 64;
static constexpr int Mrows = Bc * Tc;      // 8192

union FragBF {
  v16bf v;
  uint4 q[2];
};

__device__ inline v8f wmma_bf16(v16bf a, v16bf b, v8f c) {
  // v_wmma_f32_16x16x32_bf16 : D = A(16x32) * B(32x16) + C(16x16 f32)
  return __builtin_amdgcn_wmma_f32_16x16x32_bf16(false, a, false, b, (short)0, c,
                                                 false, false);
}

// ---------------------------------------------------------------------------
// f32 -> bf16 elementwise conversion, 4 elements per thread
// ---------------------------------------------------------------------------
__global__ void cvt_f32_bf16_v4(const float* __restrict__ in,
                                bf16* __restrict__ out, int n4) {
  int i = blockIdx.x * blockDim.x + threadIdx.x;
  if (i < n4) {
    float4 f = ((const float4*)in)[i];
    v4bf o;
    o[0] = (bf16)f.x; o[1] = (bf16)f.y; o[2] = (bf16)f.z; o[3] = (bf16)f.w;
    *(v4bf*)(out + (size_t)i * 4) = o;
  }
}

// ---------------------------------------------------------------------------
// GEMM helpers: stage one 128x32 A tile and one (transposed) 32x128 W tile.
// 256 threads; each thread owns two 16B chunks of each tile.
// ---------------------------------------------------------------------------
__device__ inline void gemm_load_regs(const bf16* __restrict__ A,
                                      const bf16* __restrict__ W,
                                      int row0, int n0, int k0, int N, int K,
                                      int tid, uint4 aR[2], uint4 wR[2]) {
#pragma unroll
  for (int c = 0; c < 2; ++c) {
    int ch  = tid * 2 + c;          // 0..511
    int r   = ch >> 2;              // row 0..127
    int off = (ch & 3) * 8;         // 0,8,16,24
    aR[c] = *(const uint4*)(A + (size_t)(row0 + r) * K + k0 + off);
  }
#pragma unroll
  for (int c = 0; c < 2; ++c) {
    int ch = tid * 2 + c;           // 0..511
    int kr = ch >> 4;               // k row 0..31
    int ng = (ch & 15) * 8;         // n group 0..120
    wR[c] = *(const uint4*)(W + (size_t)(k0 + kr) * N + n0 + ng);
  }
}

__device__ inline void gemm_store_lds(bf16* __restrict__ As,
                                      bf16* __restrict__ Wt,
                                      int tid, const uint4 aR[2],
                                      const uint4 wR[2]) {
#pragma unroll
  for (int c = 0; c < 2; ++c) {
    int ch  = tid * 2 + c;
    int r   = ch >> 2;
    int off = (ch & 3) * 8;
    *(uint4*)(&As[r * 40 + off]) = aR[c];
  }
#pragma unroll
  for (int c = 0; c < 2; ++c) {
    int ch = tid * 2 + c;
    int kr = ch >> 4;
    int ng = (ch & 15) * 8;
    const bf16* dv = (const bf16*)&wR[c];
#pragma unroll
    for (int j = 0; j < 8; ++j) Wt[(ng + j) * 40 + kr] = dv[j];
  }
}

// ---------------------------------------------------------------------------
// GEMM: C[M,N] = epilogue(A[M,K] @ W[K,N] + bias)
// Compile-time epilogue: +bias, RELU, row mask by source_len, bf16/f32 store.
// Block: 256 threads (8 waves). Tile: BM=128, BN=128, BK=32, double-buffered
// LDS with register prefetch (one barrier per K step).
// Wave (wm in 0..3, wn in 0..1) owns 32x64 = 2x4 WMMA tiles.
// ---------------------------------------------------------------------------
template <bool RELU, bool MASK, bool OUTB, bool OUTF>
__global__ __launch_bounds__(256) void gemm_bf16_wmma(
    const bf16* __restrict__ A, const bf16* __restrict__ W,
    const float* __restrict__ bias,
    bf16* __restrict__ outb, float* __restrict__ outf,
    const long long* __restrict__ slen,
    int M, int N, int K) {
  const int tid  = threadIdx.x;
  const int wid  = tid >> 5;
  const int lane = tid & 31;
  const int l16  = lane & 15;
  const int hi   = lane >> 4;
  const int kb   = hi * 8;           // fragment K base within 16-bit layout
  const int wm   = wid & 3;
  const int wn   = wid >> 2;

  __shared__ __align__(16) bf16 As[2][128 * 40];  // stride 40 (80B, 16B aligned)
  __shared__ __align__(16) bf16 Wt[2][128 * 40];  // W transposed: [n][k]

  const int row0 = blockIdx.y * 128;
  const int n0   = blockIdx.x * 128;

  // One scalar source-length per block (128-row tile never straddles batches).
  int sl = 0;
  if (MASK) sl = (int)slen[row0 >> 11];

  v8f acc[2][4];
#pragma unroll
  for (int mt = 0; mt < 2; ++mt)
#pragma unroll
    for (int nt = 0; nt < 4; ++nt) acc[mt][nt] = {};

  // prologue: stage k0 = 0 into buffer 0
  {
    uint4 aR[2], wR[2];
    gemm_load_regs(A, W, row0, n0, 0, N, K, tid, aR, wR);
    gemm_store_lds(As[0], Wt[0], tid, aR, wR);
  }
  __syncthreads();

  int cur = 0;
  for (int k0 = 0; k0 < K; k0 += 32) {
    const bool has_next = (k0 + 32) < K;
    uint4 aR[2], wR[2];
    if (has_next)  // prefetch next tile into registers (overlaps WMMA below)
      gemm_load_regs(A, W, row0, n0, k0 + 32, N, K, tid, aR, wR);

    // ---- fragments from current buffer ----
    FragBF a[2], bfr[4];
#pragma unroll
    for (int mt = 0; mt < 2; ++mt) {
      int m = wm * 32 + mt * 16 + l16;
      a[mt].q[0] = *(const uint4*)(&As[cur][m * 40 + kb]);
      a[mt].q[1] = *(const uint4*)(&As[cur][m * 40 + kb + 16]);
    }
#pragma unroll
    for (int nt = 0; nt < 4; ++nt) {
      int n = wn * 64 + nt * 16 + l16;
      bfr[nt].q[0] = *(const uint4*)(&Wt[cur][n * 40 + kb]);
      bfr[nt].q[1] = *(const uint4*)(&Wt[cur][n * 40 + kb + 16]);
    }
#pragma unroll
    for (int mt = 0; mt < 2; ++mt)
#pragma unroll
      for (int nt = 0; nt < 4; ++nt)
        acc[mt][nt] = wmma_bf16(a[mt].v, bfr[nt].v, acc[mt][nt]);

    if (has_next) {
      gemm_store_lds(As[cur ^ 1], Wt[cur ^ 1], tid, aR, wR);
      cur ^= 1;
    }
    // Single barrier per step: its s_wait_dscnt 0 also guarantees this wave's
    // reads of the old buffer completed before anyone overwrites it next step.
    __syncthreads();
  }

  // ---- branch-free epilogue ----
#pragma unroll
  for (int mt = 0; mt < 2; ++mt) {
#pragma unroll
    for (int nt = 0; nt < 4; ++nt) {
      const int gn = n0 + wn * 64 + nt * 16 + l16;
      const float bs = bias[gn];
#pragma unroll
      for (int r = 0; r < 8; ++r) {
        const int gm = row0 + wm * 32 + mt * 16 + hi * 8 + r;
        float v = acc[mt][nt][r] + bs;
        if (RELU) v = fmaxf(v, 0.0f);
        if (MASK) {
          int t = gm & (Tc - 1);
          if (t >= sl) v = 0.0f;
        }
        if (OUTB) outb[(size_t)gm * N + gn] = (bf16)v;
        if (OUTF) outf[(size_t)gm * N + gn] = v;
      }
    }
  }
}

// ---------------------------------------------------------------------------
// Flash attention. Grid: B*NH*(T/128) blocks, 256 threads (8 waves).
// Each wave owns 16 query rows; K/V tiles of 64 keys per step.
// Q/K layout [B,T,H] bf16 (head h at column h*64). ctx written bf16 same layout.
// ---------------------------------------------------------------------------
__global__ __launch_bounds__(256) void attn_wmma(
    const bf16* __restrict__ Q, const bf16* __restrict__ Kb,
    const bf16* __restrict__ Vb, const long long* __restrict__ slen,
    bf16* __restrict__ ctx) {
  const int bid = blockIdx.x;
  const int qb  = bid & 15;           // T/128 = 16
  const int h   = (bid >> 4) & 15;    // NH
  const int b   = bid >> 8;

  const int tid  = threadIdx.x;
  const int wid  = tid >> 5;
  const int lane = tid & 31;
  const int l16  = lane & 15;
  const int hi   = lane >> 4;
  const int kb   = hi * 8;

  __shared__ __align__(16) bf16 Vt[64 * 72];           // V transposed [dk][key]
  __shared__ __align__(16) bf16 Pl[8 * 16 * 72];       // per-wave P tiles
  bf16* myP = &Pl[wid * 16 * 72];

  const int qrow0 = qb * 128 + wid * 16;
  const int sl = (int)slen[b];
  const size_t rowBase = (size_t)b * Tc;

  // Q fragments (A layout): lane row = qrow0 + l16, contiguous dk slices.
  FragBF qf[2];
  {
    const bf16* qp = Q + (rowBase + qrow0 + l16) * Hc + h * DKc;
#pragma unroll
    for (int kc = 0; kc < 2; ++kc) {
      qf[kc].q[0] = *(const uint4*)(qp + kc * 32 + kb);
      qf[kc].q[1] = *(const uint4*)(qp + kc * 32 + kb + 16);
    }
  }

  v8f acc[4];
#pragma unroll
  for (int dt = 0; dt < 4; ++dt) acc[dt] = {};
  float mi[8], li[8];
#pragma unroll
  for (int r = 0; r < 8; ++r) { mi[r] = -1e30f; li[r] = 0.0f; }

  for (int kv0 = 0; kv0 < Tc; kv0 += 64) {
    // Prefetch next K/V tile (global_prefetch_b8); 4 threads cover one row.
    if (kv0 + 64 < Tc) {
      int key  = tid >> 2;            // 0..63
      int part = (tid & 3) * 16;      // 32B granules across the 128B head row
      const bf16* pk = Kb + (rowBase + kv0 + 64 + key) * Hc + h * DKc + part;
      const bf16* pv = Vb + (rowBase + kv0 + 64 + key) * Hc + h * DKc + part;
      __builtin_prefetch(pk, 0, 0);
      __builtin_prefetch(pv, 0, 0);
    }

    __syncthreads();   // previous iteration's Vt readers done
    // ---- stage V transposed: Vt[dk][key] ----
#pragma unroll
    for (int c = 0; c < 2; ++c) {
      int ch  = tid * 2 + c;          // 0..511
      int key = ch >> 3;              // 0..63
      int dg  = (ch & 7) * 8;         // 0..56
      uint4 d = *(const uint4*)(Vb + (rowBase + kv0 + key) * Hc + h * DKc + dg);
      const bf16* dv = (const bf16*)&d;
#pragma unroll
      for (int j = 0; j < 8; ++j) Vt[(dg + j) * 72 + key] = dv[j];
    }
    __syncthreads();

    // ---- scores S = Q K^T : 4 tiles of 16x16, K frag direct from global ----
    v8f s[4];
#pragma unroll
    for (int st = 0; st < 4; ++st) {
      s[st] = {};
      int key = kv0 + st * 16 + l16;
      const bf16* kp = Kb + (rowBase + key) * Hc + h * DKc;
#pragma unroll
      for (int kc = 0; kc < 2; ++kc) {
        FragBF kf;
        kf.q[0] = *(const uint4*)(kp + kc * 32 + kb);
        kf.q[1] = *(const uint4*)(kp + kc * 32 + kb + 16);
        s[st] = wmma_bf16(qf[kc].v, kf.v, s[st]);
      }
    }
    float cmask[4];
#pragma unroll
    for (int st = 0; st < 4; ++st) {
      int key = kv0 + st * 16 + l16;
      cmask[st] = (key < sl) ? 0.0f : -10000.0f;
    }

    // ---- online softmax (row = r + 8*hi, column = lane within 16) ----
    float p[4][8];
#pragma unroll
    for (int r = 0; r < 8; ++r) {
      float sv[4];
      float t0 = -1e30f;
#pragma unroll
      for (int st = 0; st < 4; ++st) {
        sv[st] = s[st][r] * 0.125f + cmask[st];   // 1/sqrt(64)
        t0 = fmaxf(t0, sv[st]);
      }
#pragma unroll
      for (int off = 1; off < 16; off <<= 1)
        t0 = fmaxf(t0, __shfl_xor(t0, off, 32));
      float nm   = fmaxf(mi[r], t0);
      float corr = __expf(mi[r] - nm);
      float rs   = 0.0f;
#pragma unroll
      for (int st = 0; st < 4; ++st) {
        p[st][r] = __expf(sv[st] - nm);
        rs += p[st][r];
      }
#pragma unroll
      for (int off = 1; off < 16; off <<= 1)
        rs += __shfl_xor(rs, off, 32);
      li[r] = li[r] * corr + rs;
      mi[r] = nm;
#pragma unroll
      for (int dt = 0; dt < 4; ++dt) acc[dt][r] *= corr;
    }

    // ---- C-layout -> A-layout for P via wave-private LDS ----
#pragma unroll
    for (int st = 0; st < 4; ++st)
#pragma unroll
      for (int r = 0; r < 8; ++r)
        myP[(r + 8 * hi) * 72 + st * 16 + l16] = (bf16)p[st][r];

    FragBF pf[2];
    {
      const bf16* pp = myP + l16 * 72;
#pragma unroll
      for (int kc = 0; kc < 2; ++kc) {
        pf[kc].q[0] = *(const uint4*)(pp + kc * 32 + kb);
        pf[kc].q[1] = *(const uint4*)(pp + kc * 32 + kb + 16);
      }
    }

    // ---- ctx += P @ V ----
#pragma unroll
    for (int dt = 0; dt < 4; ++dt) {
      int n = dt * 16 + l16;
#pragma unroll
      for (int kc = 0; kc < 2; ++kc) {
        FragBF vf;
        vf.q[0] = *(const uint4*)(&Vt[n * 72 + kc * 32 + kb]);
        vf.q[1] = *(const uint4*)(&Vt[n * 72 + kc * 32 + kb + 16]);
        acc[dt] = wmma_bf16(pf[kc].v, vf.v, acc[dt]);
      }
    }
  }

  // ---- normalize and store ctx (bf16, [B,T,H]) ----
#pragma unroll
  for (int r = 0; r < 8; ++r) {
    const float inv = 1.0f / li[r];
    const int gm = qrow0 + 8 * hi + r;
#pragma unroll
    for (int dt = 0; dt < 4; ++dt) {
      ctx[(rowBase + gm) * Hc + h * DKc + dt * 16 + l16] =
          (bf16)(acc[dt][r] * inv);
    }
  }
}

// ---------------------------------------------------------------------------
// residual + LayerNorm: out = LN(attn + hidden) * g + b. One block per row.
// ---------------------------------------------------------------------------
__global__ __launch_bounds__(256) void resid_ln(
    const float* __restrict__ attn, const float* __restrict__ hid,
    const float* __restrict__ g, const float* __restrict__ be,
    float* __restrict__ out) {
  const int row = blockIdx.x;
  const int tid = threadIdx.x;
  __shared__ float s1[256], s2[256];

  float x[4];
  float a = 0.0f, sq = 0.0f;
#pragma unroll
  for (int i = 0; i < 4; ++i) {
    int c = tid + i * 256;
    float v = attn[(size_t)row * Hc + c] + hid[(size_t)row * Hc + c];
    x[i] = v; a += v; sq += v * v;
  }
  s1[tid] = a; s2[tid] = sq;
  __syncthreads();
  for (int o = 128; o > 0; o >>= 1) {
    if (tid < o) { s1[tid] += s1[tid + o]; s2[tid] += s2[tid + o]; }
    __syncthreads();
  }
  float mu  = s1[0] * (1.0f / Hc);
  float var = s2[0] * (1.0f / Hc) - mu * mu;
  float inv = rsqrtf(var + 1e-12f);
#pragma unroll
  for (int i = 0; i < 4; ++i) {
    int c = tid + i * 256;
    out[(size_t)row * Hc + c] = (x[i] - mu) * inv * g[c] + be[c];
  }
}

// ---------------------------------------------------------------------------
// launch
// ---------------------------------------------------------------------------
extern "C" void kernel_launch(void* const* d_in, const int* in_sizes, int n_in,
                              void* d_out, int out_size, void* d_ws, size_t ws_size,
                              hipStream_t stream) {
  const float* hid_f = (const float*)d_in[0];
  const float* src_f = (const float*)d_in[1];
  const long long* slen = (const long long*)d_in[2];
  const float* Wq = (const float*)d_in[3];  const float* bq = (const float*)d_in[4];
  const float* Wk = (const float*)d_in[5];  const float* bk = (const float*)d_in[6];
  const float* Wv = (const float*)d_in[7];  const float* bv = (const float*)d_in[8];
  const float* Wd = (const float*)d_in[9];  const float* bd = (const float*)d_in[10];
  const float* Wp1 = (const float*)d_in[11]; const float* bp1 = (const float*)d_in[12];
  const float* Wp2 = (const float*)d_in[13]; const float* bp2 = (const float*)d_in[14];
  const float* ln_g = (const float*)d_in[15]; const float* ln_b = (const float*)d_in[16];
  float* outp = (float*)d_out;

  const size_t actE = (size_t)Mrows * Hc;   // 8M elems
  const size_t wE   = (size_t)Hc * Hc;      // 1M elems

  char* ws = (char*)d_ws;
  size_t off = 0;
  auto alloc = [&](size_t bytes) -> char* {
    char* p = ws + off;
    off += (bytes + 255) & ~(size_t)255;
    return p;
  };
  bf16* wq_b  = (bf16*)alloc(wE * 2);
  bf16* wk_b  = (bf16*)alloc(wE * 2);
  bf16* wv_b  = (bf16*)alloc(wE * 2);
  bf16* wd_b  = (bf16*)alloc(wE * 2);
  bf16* wp1_b = (bf16*)alloc(wE * 2);
  bf16* wp2_b = (bf16*)alloc(wE * 2);
  bf16* src_b = (bf16*)alloc(actE * 2);
  bf16* hid_b = (bf16*)alloc(actE * 2);
  bf16* p1_b  = (bf16*)alloc(actE * 2);   // reused as ctx after attention
  bf16* p_b   = (bf16*)alloc(actE * 2);
  bf16* q_b   = (bf16*)alloc(actE * 2);
  bf16* k_b   = (bf16*)alloc(actE * 2);
  bf16* v_b   = (bf16*)alloc(actE * 2);
  float* attn_f = (float*)alloc(actE * 4);
  (void)ws_size; (void)n_in; (void)in_sizes; (void)out_size;

  // ---- conversions to bf16 (vectorized x4) ----
  {
    int nW4 = (int)(wE / 4), nA4 = (int)(actE / 4);
    dim3 bw((nW4 + 255) / 256), ba((nA4 + 255) / 256);
    cvt_f32_bf16_v4<<<bw, 256, 0, stream>>>(Wq,  wq_b,  nW4);
    cvt_f32_bf16_v4<<<bw, 256, 0, stream>>>(Wk,  wk_b,  nW4);
    cvt_f32_bf16_v4<<<bw, 256, 0, stream>>>(Wv,  wv_b,  nW4);
    cvt_f32_bf16_v4<<<bw, 256, 0, stream>>>(Wd,  wd_b,  nW4);
    cvt_f32_bf16_v4<<<bw, 256, 0, stream>>>(Wp1, wp1_b, nW4);
    cvt_f32_bf16_v4<<<bw, 256, 0, stream>>>(Wp2, wp2_b, nW4);
    cvt_f32_bf16_v4<<<ba, 256, 0, stream>>>(src_f, src_b, nA4);
    cvt_f32_bf16_v4<<<ba, 256, 0, stream>>>(hid_f, hid_b, nA4);
  }

  dim3 gg(Hc / 128, Mrows / 128);   // (8, 64)

  // P1 = relu(src @ Wp1 + bp1)
  gemm_bf16_wmma<true, false, true, false><<<gg, 256, 0, stream>>>(
      src_b, wp1_b, bp1, p1_b, nullptr, nullptr, Mrows, Hc, Hc);
  // P = mask(P1 @ Wp2 + bp2)
  gemm_bf16_wmma<false, true, true, false><<<gg, 256, 0, stream>>>(
      p1_b, wp2_b, bp2, p_b, nullptr, slen, Mrows, Hc, Hc);
  // Q = hidden @ Wq + bq
  gemm_bf16_wmma<false, false, true, false><<<gg, 256, 0, stream>>>(
      hid_b, wq_b, bq, q_b, nullptr, nullptr, Mrows, Hc, Hc);
  // K = P @ Wk + bk
  gemm_bf16_wmma<false, false, true, false><<<gg, 256, 0, stream>>>(
      p_b, wk_b, bk, k_b, nullptr, nullptr, Mrows, Hc, Hc);
  // V = P @ Wv + bv
  gemm_bf16_wmma<false, false, true, false><<<gg, 256, 0, stream>>>(
      p_b, wv_b, bv, v_b, nullptr, nullptr, Mrows, Hc, Hc);

  // ctx = softmax(QK^T/8 + mask) V   (ctx reuses p1_b)
  bf16* ctx_b = p1_b;
  attn_wmma<<<Bc * NHc * (Tc / 128), 256, 0, stream>>>(q_b, k_b, v_b, slen, ctx_b);

  // attn_out = ctx @ Wd + bd (f32)
  gemm_bf16_wmma<false, false, false, true><<<gg, 256, 0, stream>>>(
      ctx_b, wd_b, bd, nullptr, attn_f, nullptr, Mrows, Hc, Hc);

  // out = LayerNorm(attn_out + hidden)
  resid_ln<<<Mrows, 256, 0, stream>>>(attn_f, hid_f, ln_g, ln_b, outp);
}
